// TemporalEncoder_52742198395125
// MI455X (gfx1250) — matrix operand order repaired and checked by reading, not compiled, not run on hardware
//
#include <hip/hip_runtime.h>
#include <math.h>

// ---------------------------------------------------------------------------
// TemporalEncoder on MI455X (gfx1250):
//   - ballot-compacted edge scatter (f32 global atomics)
//   - fused encode+gate kernel: agg->LDS(bf16) -> WMMA -> tanh -> LDS ->
//     WMMA gates -> state update.  h never touches HBM.
//   - branch-free activations: v_exp_f32 + v_rcp_f32 (no IEEE div sequences)
// ---------------------------------------------------------------------------

#define DIM 128
#define T_STEPS 8

typedef __bf16 bf16_t;
typedef __attribute__((ext_vector_type(16))) __bf16 v16bf;
typedef __attribute__((ext_vector_type(8)))  float  v8f;

struct Frag32B { uint4 lo, hi; };           // 32 raw bytes
struct BF16x8  { bf16_t v[8]; };            // 16 raw bytes

__device__ __forceinline__ v16bf frag_from(const uint4& lo, const uint4& hi) {
    Frag32B f{lo, hi};
    return __builtin_bit_cast(v16bf, f);
}

// Branch-free activations: one v_exp_f32 + one v_rcp_f32 each.
__device__ __forceinline__ float fast_rcp(float x) {
    return __builtin_amdgcn_rcpf(x);                  // v_rcp_f32, ~1 ulp
}
__device__ __forceinline__ float fast_sigmoid(float x) {
    return fast_rcp(1.0f + __expf(-x));
}
__device__ __forceinline__ float fast_tanh(float x) {
    float xc = fminf(fmaxf(x, -15.0f), 15.0f);        // avoid inf/inf
    float e  = __expf(2.0f * xc);
    return (e - 1.0f) * fast_rcp(e + 1.0f);
}

// ---------------------------------------------------------------------------
// Pack a [K x 128] f32 weight matrix into bf16 WMMA B-fragment order:
//   dst[colTile(8)][kChunk(kcTotal)][lane(32)][elem(16)]
// Per ISA 7.12.2 (B 32x16 bf16): lane l -> N = l&15, half = l>>4,
// element i -> K = 32*chunk + 8*half + (i&7) + 16*(i>>3).
// ---------------------------------------------------------------------------
__global__ void pack_weights(const float* __restrict__ W, bf16_t* __restrict__ dst,
                             int kcTotal, int kcOff) {
    int tid = blockIdx.x * blockDim.x + threadIdx.x;      // 8*4*32*16 = 16384
    if (tid >= 8 * 4 * 32 * 16) return;
    int i    = tid & 15;
    int lane = (tid >> 4) & 31;
    int c    = (tid >> 9) & 3;                            // local k-chunk (0..3)
    int ct   = (tid >> 11) & 7;                           // column tile
    int hf   = lane >> 4;
    int n    = ct * 16 + (lane & 15);
    int k    = 32 * c + 8 * hf + (i & 7) + 16 * (i >> 3);
    size_t o = (((size_t)ct * kcTotal + (c + kcOff)) * 32 + lane) * 16 + i;
    dst[o] = (bf16_t)W[(size_t)k * DIM + n];
}

// ---------------------------------------------------------------------------
// Edge scatter with ballot compaction: each wave32 scans 32 edges (coalesced),
// then the whole wave cooperates on each time-matching edge (lane owns 4 dims).
// ---------------------------------------------------------------------------
__global__ void scatter_edges(const int* __restrict__ edge_index,
                              const int* __restrict__ edge_type,
                              const int* __restrict__ edge_time,
                              const float* __restrict__ edge_weight,
                              const float* __restrict__ node_emb,
                              const float* __restrict__ rel_emb,
                              float* __restrict__ agg, int E, int t) {
    const int wave = (blockIdx.x * blockDim.x + threadIdx.x) >> 5;
    const int lane = threadIdx.x & 31;
    const int e    = wave * 32 + lane;                    // this lane's edge
    const bool inb = (e < E);
    const int  ec  = inb ? e : 0;

    const bool active = inb && (edge_time[ec] == t);
    // Preload per-lane edge data (coalesced); broadcast later via shfl.
    const float w_l   = edge_weight[ec];
    const int   src_l = edge_index[ec];
    const int   dst_l = edge_index[E + ec];
    const int   rel_l = edge_type[ec];

    unsigned int mask = (unsigned int)__ballot(active);   // wave32: low 32 bits
    while (mask) {
        const int b = __ffs(mask) - 1;
        mask &= mask - 1;
        const float w   = __shfl(w_l, b);
        const int   src = __shfl(src_l, b);
        const int   dst = __shfl(dst_l, b);
        const int   rel = __shfl(rel_l, b);
        const float4 s4 = *reinterpret_cast<const float4*>(node_emb + (size_t)src * DIM + lane * 4);
        const float4 r4 = *reinterpret_cast<const float4*>(rel_emb  + (size_t)rel * DIM + lane * 4);
        float* a = agg + (size_t)dst * DIM + lane * 4;
        unsafeAtomicAdd(a + 0, s4.x * r4.x * w);
        unsafeAtomicAdd(a + 1, s4.y * r4.y * w);
        unsafeAtomicAdd(a + 2, s4.z * r4.z * w);
        unsafeAtomicAdd(a + 3, s4.w * r4.w * w);
    }
}

// ---------------------------------------------------------------------------
// Build a 16-bit A-fragment from LDS (ISA 7.12.2, A 16x32 bf16):
// lane l: M = l&15, half = l>>4; K groups {8h..8h+7} and {16+8h..16+8h+7}
// are contiguous per lane -> two 16B LDS loads.
// ---------------------------------------------------------------------------
__device__ __forceinline__ v16bf a_frag_lds(const bf16_t* sA, int ldk, int m,
                                            int hf, int kbase) {
    int off = m * ldk + kbase + 8 * hf;
    uint4 a0 = *reinterpret_cast<const uint4*>(&sA[off]);
    uint4 a1 = *reinterpret_cast<const uint4*>(&sA[off + 16]);
    return frag_from(a0, a1);
}

// ---------------------------------------------------------------------------
// Fused: h = tanh(agg @ W_enc);  z = sig([h|s]@[Wz;Uz] + bz);
// cand = tanh([h|s]@[Wh;Uh] + bh);  s' = (1-z)*s + z*cand.
// Block = 256 threads = 8 waves, one 16-row tile; wave ct owns column tile ct.
// ---------------------------------------------------------------------------
__global__ void __launch_bounds__(256)
encode_gate(const float* __restrict__ A,   // agg
            const float* __restrict__ S,   // state (f32, read)
            const bf16_t* __restrict__ Wp, // packed W_enc  [8][4][32][16]
            const bf16_t* __restrict__ Zp, // packed [Wz;Uz] [8][8][32][16]
            const bf16_t* __restrict__ Hp, // packed [Wh;Uh] [8][8][32][16]
            const float* __restrict__ bz, const float* __restrict__ bh,
            float* __restrict__ Sout) {
    __shared__ __align__(16) bf16_t sA[16 * DIM];    // agg tile (bf16)
    __shared__ __align__(16) bf16_t sHS[16 * 256];   // [h | state] tile (bf16)

    const int tid = threadIdx.x;
    const int rowBase = blockIdx.x * 16;
    {   // stage agg -> sA and state -> sHS[:,128:256]  (8+8 elems / thread)
        int row = tid >> 4;
        int col = (tid & 15) * 8;
        const float* ga = A + (size_t)(rowBase + row) * DIM + col;
        const float* gs = S + (size_t)(rowBase + row) * DIM + col;
        float4 a0 = reinterpret_cast<const float4*>(ga)[0];
        float4 a1 = reinterpret_cast<const float4*>(ga)[1];
        float4 s0 = reinterpret_cast<const float4*>(gs)[0];
        float4 s1 = reinterpret_cast<const float4*>(gs)[1];
        BF16x8 pa = {{(bf16_t)a0.x, (bf16_t)a0.y, (bf16_t)a0.z, (bf16_t)a0.w,
                      (bf16_t)a1.x, (bf16_t)a1.y, (bf16_t)a1.z, (bf16_t)a1.w}};
        BF16x8 ps = {{(bf16_t)s0.x, (bf16_t)s0.y, (bf16_t)s0.z, (bf16_t)s0.w,
                      (bf16_t)s1.x, (bf16_t)s1.y, (bf16_t)s1.z, (bf16_t)s1.w}};
        *reinterpret_cast<uint4*>(&sA[row * DIM + col])        = __builtin_bit_cast(uint4, pa);
        *reinterpret_cast<uint4*>(&sHS[row * 256 + DIM + col]) = __builtin_bit_cast(uint4, ps);
    }
    __syncthreads();

    const int lane = tid & 31;
    const int ct   = tid >> 5;            // column tile (0..7)
    const int m    = lane & 15;
    const int hf   = lane >> 4;
    const int n    = ct * 16 + m;

    // ---- Phase 1: h = tanh(agg @ W_enc), written straight into sHS[:,0:128]
    {
        const uint4* wp4 = reinterpret_cast<const uint4*>(Wp);
        v8f acc = {};
#pragma unroll
        for (int c = 0; c < 4; ++c) {
            v16bf a = a_frag_lds(sA, DIM, m, hf, 32 * c);
            size_t bo = (((size_t)ct * 4 + c) * 32 + lane) * 2;   // uint4 units
            v16bf b = frag_from(wp4[bo], wp4[bo + 1]);
            acc = __builtin_amdgcn_wmma_f32_16x16x32_bf16(false, a, false, b,
                                                          (short)0, acc, false, false);
        }
#pragma unroll
        for (int v = 0; v < 8; ++v) {                 // C/D layout: row = 8*hf+v
            sHS[(hf * 8 + v) * 256 + n] = (bf16_t)fast_tanh(acc[v]);
        }
    }
    __syncthreads();

    // ---- Phase 2: gate GEMMs over K=256 ([h|state]) + elementwise update
    const uint4* zp4 = reinterpret_cast<const uint4*>(Zp);
    const uint4* hp4 = reinterpret_cast<const uint4*>(Hp);
    v8f accZ = {}, accC = {};
#pragma unroll
    for (int c = 0; c < 8; ++c) {
        v16bf a = a_frag_lds(sHS, 256, m, hf, 32 * c);
        size_t bo = (((size_t)ct * 8 + c) * 32 + lane) * 2;
        v16bf bzf = frag_from(zp4[bo], zp4[bo + 1]);
        v16bf bhf = frag_from(hp4[bo], hp4[bo + 1]);
        accZ = __builtin_amdgcn_wmma_f32_16x16x32_bf16(false, a, false, bzf,
                                                       (short)0, accZ, false, false);
        accC = __builtin_amdgcn_wmma_f32_16x16x32_bf16(false, a, false, bhf,
                                                       (short)0, accC, false, false);
    }
    const float bzv = bz[n], bhv = bh[n];
#pragma unroll
    for (int v = 0; v < 8; ++v) {
        int row = rowBase + hf * 8 + v;
        float s    = S[(size_t)row * DIM + n];        // f32 state for the blend
        float z    = fast_sigmoid(accZ[v] + bzv);
        float cand = fast_tanh(accC[v] + bhv);
        Sout[(size_t)row * DIM + n] = (1.0f - z) * s + z * cand;
    }
}

// ---------------------------------------------------------------------------
extern "C" void kernel_launch(void* const* d_in, const int* in_sizes, int n_in,
                              void* d_out, int out_size, void* d_ws, size_t ws_size,
                              hipStream_t stream) {
    const int*   edge_index  = (const int*)d_in[0];
    const int*   edge_type   = (const int*)d_in[1];
    const int*   edge_time   = (const int*)d_in[2];
    const float* edge_weight = (const float*)d_in[3];
    const float* node_emb    = (const float*)d_in[4];
    const float* rel_emb     = (const float*)d_in[5];
    const float* W_enc       = (const float*)d_in[6];
    const float* Wz          = (const float*)d_in[7];
    const float* Uz          = (const float*)d_in[8];
    const float* Wh          = (const float*)d_in[9];
    const float* Uh          = (const float*)d_in[10];
    const float* bz          = (const float*)d_in[11];
    const float* bh          = (const float*)d_in[12];

    const int E = in_sizes[1];                  // N_EDGES
    const int N = in_sizes[4] / DIM;            // N_NODES (20000 = 1250*16)

    // Workspace layout
    char* ws = (char*)d_ws;
    const size_t nd = (size_t)N * DIM * sizeof(float);
    float*  AG     = (float*)ws;  ws += nd;                          // agg
    float*  S      = (float*)ws;  ws += nd;                          // recurrent state
    bf16_t* Wenc_p = (bf16_t*)ws; ws += (size_t)8 * 4 * 32 * 16 * sizeof(bf16_t);
    bf16_t* Zp     = (bf16_t*)ws; ws += (size_t)8 * 8 * 32 * 16 * sizeof(bf16_t);
    bf16_t* Hp     = (bf16_t*)ws; ws += (size_t)8 * 8 * 32 * 16 * sizeof(bf16_t);

    // Pack weights into WMMA B-fragment order (once per launch, deterministic)
    pack_weights<<<64, 256, 0, stream>>>(W_enc, Wenc_p, 4, 0);
    pack_weights<<<64, 256, 0, stream>>>(Wz, Zp, 8, 0);
    pack_weights<<<64, 256, 0, stream>>>(Uz, Zp, 8, 4);
    pack_weights<<<64, 256, 0, stream>>>(Wh, Hp, 8, 0);
    pack_weights<<<64, 256, 0, stream>>>(Uh, Hp, 8, 4);
    hipMemsetAsync(S, 0, nd, stream);

    const int rowBlocks     = N / 16;
    const int nWaves        = (E + 31) / 32;                 // 32 edges / wave
    const int scatterBlocks = (nWaves * 32 + 255) / 256;
    for (int t = 0; t < T_STEPS; ++t) {
        hipMemsetAsync(AG, 0, nd, stream);
        scatter_edges<<<scatterBlocks, 256, 0, stream>>>(
            edge_index, edge_type, edge_time, edge_weight, node_emb, rel_emb, AG, E, t);
        float* out = (t == T_STEPS - 1) ? (float*)d_out : S;
        encode_gate<<<rowBlocks, 256, 0, stream>>>(AG, S, Wenc_p, Zp, Hp, bz, bh, out);
    }
}
